// ParallelLorentz_14705968022149
// MI455X (gfx1250) — matrix operand order repaired and checked by reading, not compiled, not run on hardware
//
#include <hip/hip_runtime.h>
#include <stdint.h>

// ---------------------------------------------------------------------------
// Types for CDNA5 WMMA
// ---------------------------------------------------------------------------
typedef __attribute__((ext_vector_type(16))) __bf16 v16bf;
typedef __attribute__((ext_vector_type(8)))  float  v8f;
typedef __attribute__((ext_vector_type(4)))  unsigned int v4u;

struct Frag2 { v4u lo, hi; };   // 32 bytes == one v16bf

__device__ __forceinline__ uint16_t f2bf(float f) {
    uint32_t u = __float_as_uint(f);
    uint32_t r = (u + 0x7FFFu + ((u >> 16) & 1u)) >> 16;   // RNE
    return (uint16_t)r;
}
__device__ __forceinline__ float bf2f(uint16_t h) {
    return __uint_as_float(((uint32_t)h) << 16);
}

// Low 32 bits of a flat pointer into the LDS aperture == LDS byte address
__device__ __forceinline__ unsigned lds_addr32(const void* p) {
    return (unsigned)(uintptr_t)p;
}

// ---------------------------------------------------------------------------
// Problem constants
// ---------------------------------------------------------------------------
#define PB 256          // batch
#define PS 512          // sequence
#define PF 128          // features
#define PH 1024         // hidden
#define PE 128          // embed
#define CHUNK_B 32
#define CHUNK_ROWS (CHUNK_B * PS)   // 16384
#define NCHUNK (PB / CHUNK_B)       // 8

// GEMM tiling
#define TILE_M 128
#define TILE_N 128
#define KSTEP 32
#define LDSS 40                       // LDS row stride in bf16 elements (80B, 16B aligned)
#define ABUF_ELEMS (TILE_M * LDSS)    // 5120 elements per A buffer
#define BBUF_ELEMS (TILE_N * LDSS)    // 5120 elements per B buffer

// ---------------------------------------------------------------------------
// WMMA GEMM: out[M,N] = A[M,K] * W[K,N] + bias ; W supplied transposed (Bt[N,K], bf16)
// a_f32 : A is float32 (converted to bf16 while staging to LDS), else bf16
// out_f32 : write float32, else bf16
// grid = (M/128, Ntot/128), block = 256 threads (8 waves: 2 M x 4 N)
// Double-buffered LDS; bf16 operands staged with global_load_async_to_lds_b128
// (ASYNCcnt path), prefetched one K-step ahead of the WMMA consumption.
// ---------------------------------------------------------------------------
__global__ __launch_bounds__(256)
void wmma_gemm_kernel(const void* __restrict__ Aptr, int lda, int a_f32,
                      const uint16_t* __restrict__ Bt, int Ktot,
                      const float* __restrict__ bias,
                      void* __restrict__ outp, int ldout, int out_f32)
{
    // [A buf0][A buf1][B buf0][B buf1]
    __shared__ uint16_t smem[2 * ABUF_ELEMS + 2 * BBUF_ELEMS];

    const int tid  = threadIdx.x;
    const int lane = tid & 31;
    const int wave = tid >> 5;
    const int wm   = wave >> 2;     // 0..1  (64-row slabs)
    const int wn   = wave & 3;      // 0..3  (32-col slabs)
    const int m0   = blockIdx.x * TILE_M;
    const int n0   = blockIdx.y * TILE_N;

    v8f acc[4][2] = {};

    const int r    = tid >> 1;      // 0..127 staged row
    const int half = tid & 1;       // which 16-element half of the K-slice

    // Stage one 128x32 A tile + 128x32 Bt tile into LDS buffer p for K offset k0
    auto stage = [&](int p, int k0) {
        uint16_t* As = &smem[p * ABUF_ELEMS];
        uint16_t* Bs = &smem[2 * ABUF_ELEMS + p * BBUF_ELEMS];
        if (a_f32) {
            // f32 -> bf16 conversion path (through VGPRs + ds_store)
            const float* A = (const float*)Aptr + (size_t)(m0 + r) * lda + k0 + half * 16;
            alignas(16) uint16_t tmp[16];
#pragma unroll
            for (int j = 0; j < 16; ++j) tmp[j] = f2bf(A[j]);
            *(v4u*)&As[r * LDSS + half * 16]     = *(const v4u*)&tmp[0];
            *(v4u*)&As[r * LDSS + half * 16 + 8] = *(const v4u*)&tmp[8];
        } else {
            // async DMA global -> LDS, no VGPR staging
            const uint16_t* A = (const uint16_t*)Aptr + (size_t)(m0 + r) * lda + k0 + half * 16;
            const unsigned  la = lds_addr32(&As[r * LDSS + half * 16]);
            asm volatile("global_load_async_to_lds_b128 %0, %1, off"
                         :: "v"(la),       "v"((uint64_t)(uintptr_t)A) : "memory");
            asm volatile("global_load_async_to_lds_b128 %0, %1, off"
                         :: "v"(la + 16u), "v"((uint64_t)(uintptr_t)(A + 8)) : "memory");
        }
        {
            const uint16_t* Bp = Bt + (size_t)(n0 + r) * Ktot + k0 + half * 16;
            const unsigned  lb = lds_addr32(&Bs[r * LDSS + half * 16]);
            asm volatile("global_load_async_to_lds_b128 %0, %1, off"
                         :: "v"(lb),       "v"((uint64_t)(uintptr_t)Bp) : "memory");
            asm volatile("global_load_async_to_lds_b128 %0, %1, off"
                         :: "v"(lb + 16u), "v"((uint64_t)(uintptr_t)(Bp + 8)) : "memory");
        }
    };

    const int nk = Ktot / KSTEP;
    stage(0, 0);

    const int lhalf = lane >> 4;   // 0/1
    const int lrow  = lane & 15;

    for (int ks = 0; ks < nk; ++ks) {
        // own async copies done, then workgroup-wide visibility
        asm volatile("s_wait_asynccnt 0x0" ::: "memory");
        __syncthreads();

        // prefetch next K-slab into the other buffer while computing this one
        if (ks + 1 < nk) stage((ks + 1) & 1, (ks + 1) * KSTEP);

        const int p = ks & 1;
        const uint16_t* As = &smem[p * ABUF_ELEMS];
        const uint16_t* Bs = &smem[2 * ABUF_ELEMS + p * BBUF_ELEMS];

        // A fragments: 16x32 bf16, lane half selects K groups {0-7,16-23} / {8-15,24-31}
        v16bf af[4];
#pragma unroll
        for (int mi = 0; mi < 4; ++mi) {
            const uint16_t* pp = &As[(wm * 64 + mi * 16 + lrow) * LDSS];
            Frag2 f;
            f.lo = *(const v4u*)(pp + lhalf * 8);
            f.hi = *(const v4u*)(pp + 16 + lhalf * 8);
            af[mi] = __builtin_bit_cast(v16bf, f);
        }
        // B fragments: lane half selects K groups {0-7,16-23} / {8-15,24-31}; N = lane%16
        v16bf bfr[2];
#pragma unroll
        for (int ni = 0; ni < 2; ++ni) {
            const uint16_t* pp = &Bs[(wn * 32 + ni * 16 + lrow) * LDSS];
            Frag2 f;
            f.lo = *(const v4u*)(pp + lhalf * 8);
            f.hi = *(const v4u*)(pp + 16 + lhalf * 8);
            bfr[ni] = __builtin_bit_cast(v16bf, f);
        }
#pragma unroll
        for (int mi = 0; mi < 4; ++mi)
#pragma unroll
            for (int ni = 0; ni < 2; ++ni)
                acc[mi][ni] = __builtin_amdgcn_wmma_f32_16x16x32_bf16(
                    false, af[mi], false, bfr[ni], (short)0, acc[mi][ni], false, false);
    }

    // ---- epilogue: bias add + store ----
#pragma unroll
    for (int mi = 0; mi < 4; ++mi) {
#pragma unroll
        for (int ni = 0; ni < 2; ++ni) {
            const int col = n0 + wn * 32 + ni * 16 + lrow;
            const float bv = bias ? bias[col] : 0.f;
#pragma unroll
            for (int v = 0; v < 8; ++v) {
                const int row = m0 + wm * 64 + mi * 16 + lhalf * 8 + v;
                const float val = acc[mi][ni][v] + bv;
                if (out_f32)
                    ((float*)outp)[(size_t)row * ldout + col] = val;
                else
                    ((uint16_t*)outp)[(size_t)row * ldout + col] = f2bf(val);
            }
        }
    }
}

// ---------------------------------------------------------------------------
// Weight transpose + f32->bf16 : dst[N][K] = src[K][N]
// ---------------------------------------------------------------------------
__global__ void wconv_kernel(const float* __restrict__ src, uint16_t* __restrict__ dst,
                             int K, int N)
{
    size_t i = (size_t)blockIdx.x * 256 + threadIdx.x;
    if (i >= (size_t)K * N) return;
    int n = (int)(i / K);
    int k = (int)(i % K);
    dst[i] = f2bf(src[(size_t)k * N + n]);
}

// ---------------------------------------------------------------------------
// In-place LayerNorm + exact GELU over rows of 1024 bf16 (one block per row)
// ---------------------------------------------------------------------------
__global__ __launch_bounds__(256)
void ln_gelu_kernel(uint16_t* __restrict__ act,
                    const float* __restrict__ g, const float* __restrict__ be)
{
    __shared__ float s1[256], s2[256];
    uint16_t* p = act + (size_t)blockIdx.x * PH;
    const int t = threadIdx.x;
    float x[4], s = 0.f, q = 0.f;
#pragma unroll
    for (int j = 0; j < 4; ++j) {
        x[j] = bf2f(p[t * 4 + j]);
        s += x[j]; q += x[j] * x[j];
    }
    s1[t] = s; s2[t] = q;
    __syncthreads();
    for (int off = 128; off > 0; off >>= 1) {
        if (t < off) { s1[t] += s1[t + off]; s2[t] += s2[t + off]; }
        __syncthreads();
    }
    const float mean = s1[0] * (1.f / PH);
    const float var  = s2[0] * (1.f / PH) - mean * mean;
    const float inv  = rsqrtf(var + 1e-5f);
#pragma unroll
    for (int j = 0; j < 4; ++j) {
        const int c = t * 4 + j;
        const float y  = (x[j] - mean) * inv * g[c] + be[c];
        const float ge = 0.5f * y * (1.f + erff(y * 0.70710678118654752f));
        p[c] = f2bf(ge);
    }
}

// ---------------------------------------------------------------------------
// Mean-pool over S=512 rows per batch element -> bf16 (one block per b)
// ---------------------------------------------------------------------------
__global__ __launch_bounds__(256)
void pool_kernel(const uint16_t* __restrict__ act, uint16_t* __restrict__ pooledb)
{
    const int bb = blockIdx.x;     // 0..31 within chunk
    const int t  = threadIdx.x;
#pragma unroll
    for (int j = 0; j < 4; ++j) {
        const int c = t * 4 + j;
        const uint16_t* p = act + (size_t)bb * PS * PH + c;
        float s = 0.f;
        for (int srow = 0; srow < PS; ++srow) s += bf2f(p[(size_t)srow * PH]);
        pooledb[(size_t)bb * PH + c] = f2bf(s * (1.f / PS));
    }
}

// ---------------------------------------------------------------------------
// Hyperbolic Frechet-mean finalizer: 1 block, thread b = batch element
// zbase: d_out holding z[4][256][128] f32 ; writes h[4], tangent, combined
// ---------------------------------------------------------------------------
__global__ __launch_bounds__(256)
void lorentz_kernel(const float* __restrict__ zbase,
                    const float* __restrict__ eff, const float* __restrict__ lw,
                    float* __restrict__ out)
{
    const int b = threadIdx.x;
    const float es = tanhf(eff[0]);

    float w[4];
    {
        float mx = fmaxf(fmaxf(lw[0], lw[1]), fmaxf(lw[2], lw[3]));
        float sw = 0.f;
        for (int i = 0; i < 4; ++i) { w[i] = expf(lw[i] - mx); sw += w[i]; }
        for (int i = 0; i < 4; ++i) w[i] /= sw;
    }

    float hs[4][129];
    for (int e = 0; e < 4; ++e) {
        const float* z = zbase + (size_t)e * PB * PE + (size_t)b * PE;
        float n2 = 0.f, u[128];
        for (int j = 0; j < 128; ++j) { u[j] = z[j] * es; n2 += u[j] * u[j]; }
        n2 = fmaxf(n2, 1e-8f);
        const float n  = sqrtf(n2);
        const float sc = sinhf(n) / n;
        float sp2 = 0.f;
        for (int j = 0; j < 128; ++j) { hs[e][j + 1] = sc * u[j]; sp2 += hs[e][j + 1] * hs[e][j + 1]; }
        hs[e][0] = sqrtf(1.f + sp2);            // projx
    }

    // wt = sum_i w_i * logmap0(h_i)   (time component zero)
    float wt[129];
    for (int j = 0; j < 129; ++j) wt[j] = 0.f;
    for (int e = 0; e < 4; ++e) {
        const float a0 = fmaxf(hs[e][0], 1.f + 1e-7f);
        const float d  = acoshf(a0);
        float sn2 = 0.f;
        for (int j = 1; j < 129; ++j) sn2 += hs[e][j] * hs[e][j];
        const float f = d / sqrtf(fmaxf(sn2, 1e-8f));
        for (int j = 1; j < 129; ++j) wt[j] += w[e] * f * hs[e][j];
    }

    float cur[129];
    {
        float n2 = 0.f;
        for (int j = 1; j < 129; ++j) n2 += wt[j] * wt[j];
        n2 = fmaxf(n2, 1e-8f);
        const float n  = sqrtf(n2);
        const float sc = sinhf(n) / n;
        float sp2 = 0.f;
        for (int j = 1; j < 129; ++j) { cur[j] = sc * wt[j]; sp2 += cur[j] * cur[j]; }
        cur[0] = sqrtf(1.f + sp2);
    }

    for (int it = 0; it < 10; ++it) {
        float vec[129];
        for (int j = 0; j < 129; ++j) vec[j] = 0.f;
        for (int e = 0; e < 4; ++e) {
            float dotsp = 0.f;
            for (int j = 1; j < 129; ++j) dotsp += cur[j] * hs[e][j];
            float alpha = fmaxf(cur[0] * hs[e][0] - dotsp, 1.f + 1e-7f);
            const float d = acoshf(alpha);
            float vtmp[129];
            vtmp[0] = hs[e][0] - alpha * cur[0];
            float lv = -vtmp[0] * vtmp[0];
            for (int j = 1; j < 129; ++j) {
                const float vv = hs[e][j] - alpha * cur[j];
                vtmp[j] = vv; lv += vv * vv;
            }
            const float f = w[e] * d / sqrtf(fmaxf(lv, 1e-8f));
            for (int j = 0; j < 129; ++j) vec[j] += f * vtmp[j];
        }
        // cur = projx(expmap(cur, 0.5*vec))
        float u0 = 0.5f * vec[0];
        float lu = -u0 * u0;
        for (int j = 1; j < 129; ++j) { const float uu = 0.5f * vec[j]; lu += uu * uu; }
        const float un = sqrtf(fmaxf(lu, 1e-8f));
        const float ch = coshf(un), sh = sinhf(un) / un;
        float nx[129];
        float sp2 = 0.f;
        for (int j = 1; j < 129; ++j) { nx[j] = ch * cur[j] + sh * (0.5f * vec[j]); sp2 += nx[j] * nx[j]; }
        nx[0] = sqrtf(1.f + sp2);                // projx
        for (int j = 0; j < 129; ++j) cur[j] = nx[j];
    }

    // outputs
    float* hbase = out + (size_t)4 * PB * PE;    // 131072
    for (int e = 0; e < 4; ++e) {
        float* ph = hbase + (size_t)e * PB * 129 + (size_t)b * 129;
        for (int j = 0; j < 129; ++j) ph[j] = hs[e][j];
    }
    float* pt = hbase + (size_t)4 * PB * 129 + (size_t)b * 129;  // combined_tangent
    float* pc = pt + (size_t)PB * 129;                           // combined_h
    {
        const float a0 = fmaxf(cur[0], 1.f + 1e-7f);
        const float d  = acoshf(a0);
        float sn2 = 0.f;
        for (int j = 1; j < 129; ++j) sn2 += cur[j] * cur[j];
        const float f = d / sqrtf(fmaxf(sn2, 1e-8f));
        pt[0] = 0.f;
        for (int j = 1; j < 129; ++j) pt[j] = f * cur[j];
        for (int j = 0; j < 129; ++j) pc[j] = cur[j];
    }
}

// ---------------------------------------------------------------------------
// Host orchestration
// ---------------------------------------------------------------------------
struct EmbP {
    const float *in_w, *in_b, *out_w, *out_b;
    const float *w[2], *b[2], *g[2], *be[2];
};

extern "C" void kernel_launch(void* const* d_in, const int* in_sizes, int n_in,
                              void* d_out, int out_size, void* d_ws, size_t ws_size,
                              hipStream_t stream)
{
    (void)out_size; (void)ws_size;
    const float* X[4] = { (const float*)d_in[0], (const float*)d_in[1],
                          (const float*)d_in[2], (const float*)d_in[3] };

    EmbP emb[4];
    const float* eff = nullptr;
    const float* lw  = nullptr;
    int idx = 4;
    const bool sorted = (n_in > 4 && in_sizes[4] == 1);   // jax sorted-key flattening
    if (sorted) {
        eff = (const float*)d_in[idx++];
        for (int e = 0; e < 4; ++e) {
            emb[e].in_b = (const float*)d_in[idx++];
            emb[e].in_w = (const float*)d_in[idx++];
            for (int l = 0; l < 2; ++l) {
                emb[e].b[l]  = (const float*)d_in[idx++];
                emb[e].be[l] = (const float*)d_in[idx++];
                emb[e].g[l]  = (const float*)d_in[idx++];
                emb[e].w[l]  = (const float*)d_in[idx++];
            }
            emb[e].out_b = (const float*)d_in[idx++];
            emb[e].out_w = (const float*)d_in[idx++];
        }
        lw = (const float*)d_in[idx++];
    } else {                                              // insertion-order flattening
        for (int e = 0; e < 4; ++e) {
            emb[e].in_w = (const float*)d_in[idx++];
            emb[e].in_b = (const float*)d_in[idx++];
            for (int l = 0; l < 2; ++l) {
                emb[e].w[l]  = (const float*)d_in[idx++];
                emb[e].b[l]  = (const float*)d_in[idx++];
                emb[e].g[l]  = (const float*)d_in[idx++];
                emb[e].be[l] = (const float*)d_in[idx++];
            }
            emb[e].out_w = (const float*)d_in[idx++];
            emb[e].out_b = (const float*)d_in[idx++];
        }
        eff = (const float*)d_in[idx++];
        lw  = (const float*)d_in[idx++];
    }

    // ---- carve workspace (~84 MB; activations chunked to stay inside L2) ----
    uint8_t* ws = (uint8_t*)d_ws;
    size_t off = 0;
    auto carve = [&](size_t bytes) -> uint8_t* {
        uint8_t* p = ws + off;
        off = (off + bytes + 255) & ~(size_t)255;
        return p;
    };
    uint16_t *WtIn[4], *WtL[4][2], *WtOut[4], *Pb[4];
    for (int e = 0; e < 4; ++e) WtIn[e] = (uint16_t*)carve((size_t)PH * PF * 2);
    for (int e = 0; e < 4; ++e)
        for (int l = 0; l < 2; ++l) WtL[e][l] = (uint16_t*)carve((size_t)PH * PH * 2);
    for (int e = 0; e < 4; ++e) WtOut[e] = (uint16_t*)carve((size_t)PE * PH * 2);
    for (int e = 0; e < 4; ++e) Pb[e]    = (uint16_t*)carve((size_t)PB * PH * 2);
    uint16_t* bufA = (uint16_t*)carve((size_t)CHUNK_ROWS * PH * 2);
    uint16_t* bufB = (uint16_t*)carve((size_t)CHUNK_ROWS * PH * 2);

    const dim3 blk(256);

    // ---- transpose + convert all weights to bf16 (Bt[N][K]) ----
    for (int e = 0; e < 4; ++e) {
        wconv_kernel<<<(PF * PH + 255) / 256, blk, 0, stream>>>(emb[e].in_w, WtIn[e], PF, PH);
        for (int l = 0; l < 2; ++l)
            wconv_kernel<<<(PH * PH + 255) / 256, blk, 0, stream>>>(emb[e].w[l], WtL[e][l], PH, PH);
        wconv_kernel<<<(PH * PE + 255) / 256, blk, 0, stream>>>(emb[e].out_w, WtOut[e], PH, PE);
    }

    // ---- chunked MLP pipeline ----
    const dim3 ggrid(CHUNK_ROWS / TILE_M, PH / TILE_N);   // (128, 8)
    for (int e = 0; e < 4; ++e) {
        for (int c = 0; c < NCHUNK; ++c) {
            const float* xin = X[e] + (size_t)c * CHUNK_ROWS * PF;
            // in-proj (f32 A converted on the fly) + bias
            wmma_gemm_kernel<<<ggrid, blk, 0, stream>>>(xin, PF, 1, WtIn[e], PF,
                                                        emb[e].in_b, bufA, PH, 0);
            // layer 0
            wmma_gemm_kernel<<<ggrid, blk, 0, stream>>>(bufA, PH, 0, WtL[e][0], PH,
                                                        emb[e].b[0], bufB, PH, 0);
            ln_gelu_kernel<<<CHUNK_ROWS, blk, 0, stream>>>(bufB, emb[e].g[0], emb[e].be[0]);
            // layer 1
            wmma_gemm_kernel<<<ggrid, blk, 0, stream>>>(bufB, PH, 0, WtL[e][1], PH,
                                                        emb[e].b[1], bufA, PH, 0);
            ln_gelu_kernel<<<CHUNK_ROWS, blk, 0, stream>>>(bufA, emb[e].g[1], emb[e].be[1]);
            // mean pool over S
            pool_kernel<<<CHUNK_B, blk, 0, stream>>>(bufA, Pb[e] + (size_t)c * CHUNK_B * PH);
        }
        // out-projection: [256,1024] @ [1024,128] + bias -> z (f32, straight into d_out)
        wmma_gemm_kernel<<<dim3(PB / TILE_M, PE / TILE_N), blk, 0, stream>>>(
            Pb[e], PH, 0, WtOut[e], PH, emb[e].out_b,
            (float*)d_out + (size_t)e * PB * PE, PE, 1);
    }

    // ---- hyperbolic combine ----
    lorentz_kernel<<<1, blk, 0, stream>>>((const float*)d_out, eff, lw, (float*)d_out);
}
